// Decouple_10058813407530
// MI455X (gfx1250) — matrix operand order repaired and checked by reading, not compiled, not run on hardware
//
#include <hip/hip_runtime.h>
#include <math.h>
#include <stdint.h>

#define NN 1600
#define LL 8
#define HH 64
#define KK 8
#define DD 2                // diff_step
#define CH (HH * (DD + 1))  // 192

typedef _Float16 half8  __attribute__((ext_vector_type(8)));
typedef _Float16 half16 __attribute__((ext_vector_type(16)));
typedef float    f32x8  __attribute__((ext_vector_type(8)));

__device__ __forceinline__ half16 cat8(half8 lo, half8 hi) {
  return __builtin_shufflevector(lo, hi, 0,1,2,3,4,5,6,7,8,9,10,11,12,13,14,15);
}

// A operand: row-major [M x K] f16, tile rows m0..m0+15, K slice kk..kk+31.
// lanes 0-15 row=lane, K kk+0..15;
// lanes 16-31 row=lane-16, K kk+16..31 (contiguous, matching B).
__device__ __forceinline__ half16 load_a16(const _Float16* __restrict__ base, int lda,
                                           int m0, int kk, int lane) {
  int r = lane & 15, hf = lane >> 4;
  const _Float16* p = base + (size_t)(m0 + r) * lda + kk + hf * 16;
  half8 lo = *(const half8*)p;
  half8 hi = *(const half8*)(p + 8);
  return cat8(lo, hi);
}

// B operand from BT = row-major [N x K]: lanes 0-15 col=lane K kk..kk+15;
// lanes 16-31 col=lane-16 K kk+16..kk+31.
__device__ __forceinline__ half16 load_bT16(const _Float16* __restrict__ baseT, int ldb,
                                            int n0, int kk, int lane) {
  int c = lane & 15, hf = lane >> 4;
  const _Float16* p = baseT + (size_t)(n0 + c) * ldb + kk + hf * 16;
  half8 lo = *(const half8*)p;
  half8 hi = *(const half8*)(p + 8);
  return cat8(lo, hi);
}

// ---- CDNA5 async global->LDS copy (ASYNCcnt path, ISA ch.10/15.18) ----
__device__ __forceinline__ void async_copy_b128(uint32_t lds_off, const _Float16* gptr) {
  asm volatile("global_load_async_to_lds_b128 %0, %1, off"
               :: "v"(lds_off), "v"((unsigned long long)(uintptr_t)gptr)
               : "memory");
}
__device__ __forceinline__ void async_wait_all() {
  asm volatile("s_wait_asynccnt 0" ::: "memory");
}

// -------------------- small VALU kernels --------------------

__global__ void ev_softmax_kernel(const float* __restrict__ E1, const float* __restrict__ E2,
                                  float* __restrict__ ev) {
  __shared__ float red[HH];
  int n = blockIdx.x, h = threadIdx.x;
  float s = 0.f;
#pragma unroll
  for (int f = 0; f < 16; ++f) s += E1[n * 16 + f] * E2[f * HH + h];
  red[h] = s; __syncthreads();
  for (int st = 32; st; st >>= 1) { if (h < st) red[h] = fmaxf(red[h], red[h + st]); __syncthreads(); }
  float mx = red[0]; __syncthreads();
  float e = expf(s - mx);
  red[h] = e; __syncthreads();
  for (int st = 32; st; st >>= 1) { if (h < st) red[h] += red[h + st]; __syncthreads(); }
  ev[(size_t)n * HH + h] = e / red[0];
}

__global__ void f32_to_f16_kernel(const float* __restrict__ src, _Float16* __restrict__ dst, int n) {
  int i = blockIdx.x * blockDim.x + threadIdx.x;
  if (i < n) dst[i] = (_Float16)src[i];
}

// per (l,n) row: softmax stats + top-8 (softmax is monotone so topk of raw row
// gives the same indices) + emb = row @ fc_w^T + fc_b
__global__ void rowprep_kernel(const float* __restrict__ graph,
                               const float* __restrict__ fc_w, const float* __restrict__ fc_b,
                               float* __restrict__ emb, float* __restrict__ point,
                               int* __restrict__ neighbor) {
  __shared__ float srow[NN];
  __shared__ float red[256];
  __shared__ int   sidx[256];
  __shared__ float pickv[KK];
  __shared__ int   picki[KK];
  int b = blockIdx.x, tid = threadIdx.x;
  const float* row = graph + (size_t)b * NN;
  float lm = -3.4e38f;
  for (int i = tid; i < NN; i += 256) { float x = row[i]; srow[i] = x; lm = fmaxf(lm, x); }
  red[tid] = lm; __syncthreads();
  for (int st = 128; st; st >>= 1) { if (tid < st) red[tid] = fmaxf(red[tid], red[tid + st]); __syncthreads(); }
  float rmax = red[0]; __syncthreads();
  float ls = 0.f;
  for (int i = tid; i < NN; i += 256) ls += expf(srow[i] - rmax);
  red[tid] = ls; __syncthreads();
  for (int st = 128; st; st >>= 1) { if (tid < st) red[tid] += red[tid + st]; __syncthreads(); }
  float Z = red[0]; __syncthreads();
  for (int j = 0; j < KK; ++j) {
    float bv = -3.4e38f; int bi = -1;
    for (int i = tid; i < NN; i += 256) {
      bool skip = false;
      for (int t = 0; t < j; ++t) skip |= (picki[t] == i);
      float x = srow[i];
      if (!skip && (x > bv || (x == bv && (bi < 0 || i < bi)))) { bv = x; bi = i; }
    }
    red[tid] = bv; sidx[tid] = bi; __syncthreads();
    for (int st = 128; st; st >>= 1) {
      if (tid < st) {
        float ov = red[tid + st]; int oi = sidx[tid + st];
        if (oi >= 0 && (ov > red[tid] || (ov == red[tid] && (sidx[tid] < 0 || oi < sidx[tid])))) {
          red[tid] = ov; sidx[tid] = oi;
        }
      }
      __syncthreads();
    }
    if (tid == 0) { picki[j] = sidx[0]; pickv[j] = red[0]; }
    __syncthreads();
  }
  if (tid < KK) {
    point[(size_t)b * KK + tid] = expf(pickv[tid] - rmax) / Z;
    neighbor[(size_t)b * KK + tid] = picki[tid];
  }
  float acc[KK];
#pragma unroll
  for (int k = 0; k < KK; ++k) acc[k] = 0.f;
  for (int i = tid; i < NN; i += 256) {
    float x = srow[i];
#pragma unroll
    for (int k = 0; k < KK; ++k) acc[k] += x * fc_w[k * NN + i];
  }
  for (int k = 0; k < KK; ++k) {
    __syncthreads();
    red[tid] = acc[k]; __syncthreads();
    for (int st = 128; st; st >>= 1) { if (tid < st) red[tid] += red[tid + st]; __syncthreads(); }
    if (tid == 0) emb[(size_t)b * KK + k] = red[0] + fc_b[k];
  }
}

__global__ void neighprop_kernel(const float* __restrict__ emb, const float* __restrict__ point,
                                 const int* __restrict__ neighbor, float* __restrict__ emb2) {
  int t = blockIdx.x * blockDim.x + threadIdx.x;  // flat (l,n)
  if (t >= LL * NN) return;
  int l = t / NN;
  const float* embL = emb + (size_t)l * NN * KK;
  float out[KK];
#pragma unroll
  for (int k = 0; k < KK; ++k) out[k] = emb[(size_t)t * KK + k];
#pragma unroll
  for (int j = 0; j < KK; ++j) {
    float p = point[(size_t)t * KK + j];
    int nb = neighbor[(size_t)t * KK + j];
    const float* er = embL + (size_t)nb * KK;
#pragma unroll
    for (int k = 0; k < KK; ++k) out[k] += p * er[k];
  }
#pragma unroll
  for (int k = 0; k < KK; ++k) emb2[(size_t)t * KK + k] = out[k];
}

__device__ __forceinline__ float block_ln64(float x, const float* g, const float* bb,
                                            int h, float* red) {
  red[h] = x; __syncthreads();
  for (int st = 32; st; st >>= 1) { if (h < st) red[h] += red[h + st]; __syncthreads(); }
  float mu = red[0] * (1.f / HH); __syncthreads();
  float d = x - mu;
  red[h] = d * d; __syncthreads();
  for (int st = 32; st; st >>= 1) { if (h < st) red[h] += red[h + st]; __syncthreads(); }
  float var = red[0] * (1.f / HH); __syncthreads();
  return d * rsqrtf(var + 1e-5f) * g[h] + bb[h];
}

__global__ void qkv_kernel(const float* __restrict__ emb2,
                           const float* __restrict__ qw, const float* __restrict__ qb,
                           const float* __restrict__ kw, const float* __restrict__ kb,
                           const float* __restrict__ vw, const float* __restrict__ vb,
                           const float* __restrict__ g1, const float* __restrict__ b1,
                           const float* __restrict__ g2, const float* __restrict__ b2,
                           const float* __restrict__ g3, const float* __restrict__ b3,
                           _Float16* __restrict__ qh, _Float16* __restrict__ kh,
                           float* __restrict__ v) {
  __shared__ float se[KK];
  __shared__ float red[HH];
  int b = blockIdx.x, h = threadIdx.x;
  if (h < KK) se[h] = emb2[(size_t)b * KK + h];
  __syncthreads();
  float aq = qb[h], ak = kb[h], av = vb[h];
#pragma unroll
  for (int k = 0; k < KK; ++k) {
    float e = se[k];
    aq += e * qw[h * KK + k];
    ak += e * kw[h * KK + k];
    av += e * vw[h * KK + k];
  }
  float qn = block_ln64(aq, g1, b1, h, red);
  float kn = block_ln64(ak, g2, b2, h, red);
  float vn = block_ln64(av, g3, b3, h, red);
  qh[(size_t)b * HH + h] = (_Float16)qn;
  kh[(size_t)b * HH + h] = (_Float16)kn;
  v[(size_t)b * HH + h] = vn;
}

__global__ void dual_softmax_kernel(const float* __restrict__ scores,
                                    _Float16* __restrict__ mI, _Float16* __restrict__ mV) {
  __shared__ float sr[NN];
  __shared__ float red[256];
  int i = blockIdx.x, tid = threadIdx.x;
  const float* row = scores + (size_t)i * NN;
  float lmax = -3.4e38f, lmin = 3.4e38f;
  for (int j = tid; j < NN; j += 256) {
    float x = row[j]; sr[j] = x;
    lmax = fmaxf(lmax, x); lmin = fminf(lmin, x);
  }
  red[tid] = lmax; __syncthreads();
  for (int st = 128; st; st >>= 1) { if (tid < st) red[tid] = fmaxf(red[tid], red[tid + st]); __syncthreads(); }
  float rmax = red[0]; __syncthreads();
  red[tid] = lmin; __syncthreads();
  for (int st = 128; st; st >>= 1) { if (tid < st) red[tid] = fminf(red[tid], red[tid + st]); __syncthreads(); }
  float rmin = red[0]; __syncthreads();
  float s1 = 0.f, s2 = 0.f;
  for (int j = tid; j < NN; j += 256) { s1 += expf(sr[j] - rmax); s2 += expf(rmin - sr[j]); }
  red[tid] = s1; __syncthreads();
  for (int st = 128; st; st >>= 1) { if (tid < st) red[tid] += red[tid + st]; __syncthreads(); }
  float S1 = red[0]; __syncthreads();
  red[tid] = s2; __syncthreads();
  for (int st = 128; st; st >>= 1) { if (tid < st) red[tid] += red[tid + st]; __syncthreads(); }
  float S2 = red[0]; __syncthreads();
  float i1 = 1.f / S1, i2 = 1.f / S2;
  for (int j = tid; j < NN; j += 256) {
    float x = sr[j];
    mI[(size_t)i * NN + j] = (_Float16)(expf(x - rmax) * i1);
    mV[(size_t)i * NN + j] = (_Float16)(expf(rmin - x) * i2);
  }
}

__global__ void prep_x0_kernel(const float* __restrict__ v, const float* __restrict__ ev,
                               int use_ev, float* __restrict__ x0f,
                               _Float16* __restrict__ x0T, _Float16* __restrict__ xcat) {
  int t = blockIdx.x * blockDim.x + threadIdx.x;
  if (t >= NN * HH) return;
  int n = t / HH, h = t % HH;
  float x = v[t];
  if (use_ev) x *= ev[t];
  x0f[t] = x;
  x0T[(size_t)h * NN + n] = (_Float16)x;
  xcat[(size_t)n * CH + h * (DD + 1) + 0] = (_Float16)x;
}

// -------------------- WMMA GEMM kernels --------------------

// scores[m,n] = (q[m,:] . k[n,:]) / sqrt(N). 4 waves/block, 16x16 tile each.
__global__ void scores_wmma_kernel(const _Float16* __restrict__ q, const _Float16* __restrict__ k,
                                   float* __restrict__ scores) {
  int lane = threadIdx.x & 31;
  int wave = threadIdx.x >> 5;
  int m0 = (blockIdx.x * 4 + wave) * 16;
  int n0 = blockIdx.y * 16;
  f32x8 acc = {};
#pragma unroll
  for (int kk = 0; kk < HH; kk += 32) {
    half16 a = load_a16(q, HH, m0, kk, lane);
    half16 b = load_bT16(k, HH, n0, kk, lane);
    acc = __builtin_amdgcn_wmma_f32_16x16x32_f16(false, a, false, b, (short)0, acc, false, false);
  }
  int n = n0 + (lane & 15);
  int mr = m0 + (lane >> 4) * 8;
  const float scale = 0.025f;  // 1/sqrt(1600)
#pragma unroll
  for (int r = 0; r < 8; ++r)
    scores[(size_t)(mr + r) * NN + n] = acc[r] * scale;
}

// y = alpha*(m @ x) + beta*x0f. One wave -> 16x64 strip (4 accumulators), so the
// big operand m is streamed exactly once per launch. The shared B tile (64x32 of
// xT) is double-buffered in LDS via CDNA5 async global->LDS loads, overlapped
// with the 4 WMMAs of the current K-step.
__global__ void diff_wmma_kernel(const _Float16* __restrict__ mmat, const _Float16* __restrict__ xT,
                                 const float* __restrict__ x0f, float alpha, float beta,
                                 _Float16* __restrict__ yT, _Float16* __restrict__ xcat, int dcol) {
  __shared__ _Float16 sB[2][HH * 32];  // [buf][h*32 + (K-kk)]
  int tid = threadIdx.x;
  int lane = tid & 31, wave = tid >> 5;
  int m0 = (blockIdx.x * 4 + wave) * 16;
  // staging: thread -> row h = tid>>1, 16-half chunk = tid&1 (128 threads = 4KB tile)
  int sh = tid >> 1, shalf = tid & 1;
  const _Float16* gsrc = xT + (size_t)sh * NN + shalf * 16;
  uint32_t lds0 = (uint32_t)(uintptr_t)&sB[0][sh * 32 + shalf * 16];
  uint32_t lds1 = (uint32_t)(uintptr_t)&sB[1][sh * 32 + shalf * 16];

  f32x8 acc[4] = {};
  async_copy_b128(lds0, gsrc);            // stage kt=0
  async_wait_all();
  __syncthreads();
  for (int kt = 0; kt < NN / 32; ++kt) {
    int kk = kt * 32;
    int cur = kt & 1;
    if (kt + 1 < NN / 32)                 // stage kt+1 into the other buffer
      async_copy_b128(cur ? lds0 : lds1, gsrc + kk + 32);
    __builtin_prefetch(mmat + (size_t)(m0 + (lane & 15)) * NN + kk + 64, 0, 1);
    half16 a = load_a16(mmat, NN, m0, kk, lane);
#pragma unroll
    for (int ht = 0; ht < 4; ++ht) {
      const _Float16* p = &sB[cur][(ht * 16 + (lane & 15)) * 32 + (lane >> 4) * 16];
      half8 lo = *(const half8*)p;
      half8 hi = *(const half8*)(p + 8);
      half16 b = cat8(lo, hi);
      acc[ht] = __builtin_amdgcn_wmma_f32_16x16x32_f16(false, a, false, b, (short)0,
                                                       acc[ht], false, false);
    }
    async_wait_all();                     // own wave's stage done...
    __syncthreads();                      // ...and visible to all waves
  }
  int c = lane & 15;
  int mr = m0 + (lane >> 4) * 8;
#pragma unroll
  for (int ht = 0; ht < 4; ++ht) {
    int h = ht * 16 + c;
#pragma unroll
    for (int r = 0; r < 8; ++r) {
      int i = mr + r;
      float y = alpha * acc[ht][r] + beta * x0f[(size_t)i * HH + h];
      if (yT) yT[(size_t)h * NN + i] = (_Float16)y;
      xcat[(size_t)i * CH + h * (DD + 1) + dcol] = (_Float16)y;
    }
  }
}

// Z[m,n] = Xcat[m,:] . g_w[n,:] + g_b[n]; one wave -> 16x64 strip (A reused 4x).
__global__ void proj_wmma_kernel(const _Float16* __restrict__ xcat, const _Float16* __restrict__ gwh,
                                 const float* __restrict__ gb, float* __restrict__ Z) {
  int lane = threadIdx.x & 31;
  int wave = threadIdx.x >> 5;
  int m0 = (blockIdx.x * 4 + wave) * 16;
  int n0 = blockIdx.y * 64;
  f32x8 acc[4] = {};
#pragma unroll
  for (int kk = 0; kk < CH; kk += 32) {
    half16 a = load_a16(xcat, CH, m0, kk, lane);
#pragma unroll
    for (int nt = 0; nt < 4; ++nt) {
      half16 b = load_bT16(gwh, CH, n0 + nt * 16, kk, lane);
      acc[nt] = __builtin_amdgcn_wmma_f32_16x16x32_f16(false, a, false, b, (short)0,
                                                       acc[nt], false, false);
    }
  }
  int c = lane & 15;
  int mr = m0 + (lane >> 4) * 8;
#pragma unroll
  for (int nt = 0; nt < 4; ++nt) {
    int n = n0 + nt * 16 + c;
    float bias = gb[n];
#pragma unroll
    for (int r = 0; r < 8; ++r)
      Z[(size_t)(mr + r) * NN + n] = acc[nt][r] + bias;
  }
}

// -------------------- host --------------------

extern "C" void kernel_launch(void* const* d_in, const int* in_sizes, int n_in,
                              void* d_out, int out_size, void* d_ws, size_t ws_size,
                              hipStream_t stream) {
  (void)in_sizes; (void)n_in; (void)out_size; (void)ws_size;
  const float* graph = (const float*)d_in[0];
  const float* fc_w  = (const float*)d_in[1];
  const float* fc_b  = (const float*)d_in[2];
  const float* q_w   = (const float*)d_in[3];
  const float* q_b   = (const float*)d_in[4];
  const float* k_w   = (const float*)d_in[5];
  const float* k_b   = (const float*)d_in[6];
  const float* v_w   = (const float*)d_in[7];
  const float* v_b   = (const float*)d_in[8];
  const float* ln1g  = (const float*)d_in[9];
  const float* ln1b  = (const float*)d_in[10];
  const float* ln2g  = (const float*)d_in[11];
  const float* ln2b  = (const float*)d_in[12];
  const float* ln3g  = (const float*)d_in[13];
  const float* ln3b  = (const float*)d_in[14];
  const float* E1    = (const float*)d_in[15];
  const float* E2    = (const float*)d_in[16];
  const float* g_w   = (const float*)d_in[17];
  const float* g_b   = (const float*)d_in[18];
  float* z_out = (float*)d_out;

  char* w = (char*)d_ws;
  auto bump = [&](size_t bytes) -> char* {
    char* p = w;
    w += (bytes + 255) & ~(size_t)255;
    return p;
  };
  float*    emb    = (float*)bump((size_t)LL * NN * KK * sizeof(float));
  float*    point  = (float*)bump((size_t)LL * NN * KK * sizeof(float));
  int*      nb     = (int*)bump((size_t)LL * NN * KK * sizeof(int));
  float*    emb2   = (float*)bump((size_t)LL * NN * KK * sizeof(float));
  _Float16* qh     = (_Float16*)bump((size_t)LL * NN * HH * 2);
  _Float16* kh     = (_Float16*)bump((size_t)LL * NN * HH * 2);
  float*    vbuf   = (float*)bump((size_t)LL * NN * HH * sizeof(float));
  float*    ev     = (float*)bump((size_t)NN * HH * sizeof(float));
  _Float16* gwh    = (_Float16*)bump((size_t)NN * CH * 2);
  float*    scores = (float*)bump((size_t)NN * NN * sizeof(float));
  _Float16* mI     = (_Float16*)bump((size_t)NN * NN * 2);
  _Float16* mV     = (_Float16*)bump((size_t)NN * NN * 2);
  float*    x0f    = (float*)bump((size_t)NN * HH * sizeof(float));
  _Float16* x0T    = (_Float16*)bump((size_t)NN * HH * 2);
  _Float16* x1T    = (_Float16*)bump((size_t)NN * HH * 2);
  _Float16* xcat   = (_Float16*)bump((size_t)NN * CH * 2);
  // ~31.4 MB workspace; per-l buffers reused across the l-loop so the hot
  // diffusion operands (m, xT) stay resident in the 192 MB L2.

  ev_softmax_kernel<<<NN, HH, 0, stream>>>(E1, E2, ev);
  f32_to_f16_kernel<<<(NN * CH + 255) / 256, 256, 0, stream>>>(g_w, gwh, NN * CH);
  rowprep_kernel<<<LL * NN, 256, 0, stream>>>(graph, fc_w, fc_b, emb, point, nb);
  neighprop_kernel<<<(LL * NN + 255) / 256, 256, 0, stream>>>(emb, point, nb, emb2);
  qkv_kernel<<<LL * NN, HH, 0, stream>>>(emb2, q_w, q_b, k_w, k_b, v_w, v_b,
                                         ln1g, ln1b, ln2g, ln2b, ln3g, ln3b,
                                         qh, kh, vbuf);

  const size_t plane = (size_t)NN * NN;
  for (int l = 0; l < LL; ++l) {
    const _Float16* ql = qh + (size_t)l * NN * HH;
    const _Float16* kl = kh + (size_t)l * NN * HH;
    scores_wmma_kernel<<<dim3(NN / 64, NN / 16), 128, 0, stream>>>(ql, kl, scores);
    dual_softmax_kernel<<<NN, 256, 0, stream>>>(scores, mI, mV);
    for (int var = 0; var < 2; ++var) {
      // var==0: z_V (m_V, v*ev) -> first half of d_out; var==1: z_I (m_I, v)
      const _Float16* mmat = (var == 0) ? mV : mI;
      float* Z = z_out + (var == 0 ? (size_t)0 : (size_t)LL * plane) + (size_t)l * plane;
      prep_x0_kernel<<<(NN * HH + 255) / 256, 256, 0, stream>>>(
          vbuf + (size_t)l * NN * HH, ev, (var == 0) ? 1 : 0, x0f, x0T, xcat);
      diff_wmma_kernel<<<dim3(NN / 64), 128, 0, stream>>>(
          mmat, x0T, x0f, 1.f, 0.f, x1T, xcat, 1);
      diff_wmma_kernel<<<dim3(NN / 64), 128, 0, stream>>>(
          mmat, x1T, x0f, 2.f, -1.f, (_Float16*)nullptr, xcat, 2);
      proj_wmma_kernel<<<dim3(NN / 64, NN / 64), 128, 0, stream>>>(xcat, gwh, g_b, Z);
    }
  }
}
